// MultiHeadSelfAttention_60266981097990
// MI455X (gfx1250) — matrix operand order repaired and checked by reading, not compiled, not run on hardware
//
#include <hip/hip_runtime.h>
#include <stdint.h>

// MI455X (gfx1250) fused multi-head self-attention, bf16 WMMA path
// + async global->LDS staging (ASYNCcnt) for pure-copy tiles.
// B=4, N=2048, C=1024, H=16, HD=64.

typedef __bf16 bf16;
typedef __attribute__((ext_vector_type(16))) __bf16 v16bf;
typedef __attribute__((ext_vector_type(8)))  float  v8f;

#define B_   4
#define N_   2048
#define C_   1024
#define H_   16
#define HD_  64
#define SCALE_ 0.125f   // 64^-0.5

__device__ __forceinline__ bf16 f2bf(float f) { return (bf16)f; }

// Per-lane async copy of 16 bytes global -> LDS (tracked by ASYNCcnt).
// Low 32 bits of a generic pointer into LDS are the LDS byte address.
__device__ __forceinline__ void async_copy_b128(void* lds_dst,
                                                const void* gsrc) {
  uint32_t dst = (uint32_t)(uintptr_t)lds_dst;
  asm volatile("global_load_async_to_lds_b128 %0, %1, off"
               :: "v"(dst), "v"(gsrc)
               : "memory");
}
__device__ __forceinline__ void async_wait0() {
  asm volatile("s_wait_asynccnt 0x0" ::: "memory");
}

// A-fragment (16x32 bf16, MxK) from LDS tile with row stride lda.
// lanes 0-15 hold M=lane, K=0..15 ; lanes 16-31 same M, K=16..31;
// VGPR v: K = 2v + 16*half (matches frag_b).
__device__ __forceinline__ v16bf frag_a(const bf16* sm, int lda) {
  const int lane = threadIdx.x & 31;
  const int m  = lane & 15;
  const int hp = lane >> 4;
  const bf16* row = sm + m * lda;
  v16bf a;
#pragma unroll
  for (int v = 0; v < 8; ++v) {
    const int kb = 2 * v + 16 * hp;
    a[2 * v]     = row[kb];
    a[2 * v + 1] = row[kb + 1];
  }
  return a;
}

// B-fragment (32x16 bf16, KxN) from LDS tile stored [n][k] with row stride ldb.
// lanes 0-15: N=lane, K=0..15 ; lanes 16-31: N=lane-16, K=16..31.
__device__ __forceinline__ v16bf frag_b(const bf16* sm, int ldb) {
  const int lane = threadIdx.x & 31;
  const int n  = lane & 15;
  const int kh = lane >> 4;
  const bf16* row = sm + n * ldb + kh * 16;
  v16bf b;
#pragma unroll
  for (int v = 0; v < 8; ++v) {
    b[2 * v]     = row[2 * v];
    b[2 * v + 1] = row[2 * v + 1];
  }
  return b;
}

__device__ __forceinline__ v8f wmma_bf16(v16bf a, v16bf b, v8f c) {
  return __builtin_amdgcn_wmma_f32_16x16x32_bf16(false, a, false, b, (short)0, c,
                                                 false, false);
}

// ---------------------------------------------------------------------------
// Kernel 1: QKV projection. x[8192,1024] @ w_qkv[3072,1024]^T -> Q/K/V bf16
// in [B][H][N][HD] layout. 128x128 block tile, 8 waves (2x4), 4x2 tiles/wave.
// (f32 -> bf16 conversion happens in the staging step, so loads stay on the
//  regular VMEM path here.)
// ---------------------------------------------------------------------------
__global__ __launch_bounds__(256)
void qkv_gemm(const float* __restrict__ x, const float* __restrict__ w,
              bf16* __restrict__ qws, bf16* __restrict__ kws,
              bf16* __restrict__ vws) {
  __shared__ bf16 As[128 * 32];
  __shared__ bf16 Bs[128 * 32];
  const int tid  = threadIdx.x;
  const int wave = tid >> 5;
  const int lane = tid & 31;
  const int wr = wave >> 2;          // 0..1  (64 rows each)
  const int wc = wave & 3;           // 0..3  (32 cols each)
  const int m0 = blockIdx.y * 128;   // rows of x (b*N+n)
  const int d0 = blockIdx.x * 128;   // qkv output dim (0..3071)

  v8f acc[4][2] = {};

  for (int k0 = 0; k0 < C_; k0 += 32) {
#pragma unroll
    for (int i = 0; i < 4; ++i) {
      int idx = tid + i * 256;       // 0..1023
      int row = idx >> 3;
      int q4  = (idx & 7) * 4;
      float4 va = *(const float4*)(x + (size_t)(m0 + row) * C_ + k0 + q4);
      bf16* da = As + row * 32 + q4;
      da[0] = f2bf(va.x); da[1] = f2bf(va.y);
      da[2] = f2bf(va.z); da[3] = f2bf(va.w);
      float4 vb = *(const float4*)(w + (size_t)(d0 + row) * C_ + k0 + q4);
      bf16* db = Bs + row * 32 + q4;
      db[0] = f2bf(vb.x); db[1] = f2bf(vb.y);
      db[2] = f2bf(vb.z); db[3] = f2bf(vb.w);
    }
    __syncthreads();
    v16bf af[4];
#pragma unroll
    for (int i = 0; i < 4; ++i)
      af[i] = frag_a(As + (wr * 64 + i * 16) * 32, 32);
#pragma unroll
    for (int j = 0; j < 2; ++j) {
      v16bf bw = frag_b(Bs + (wc * 32 + j * 16) * 32, 32);
#pragma unroll
      for (int i = 0; i < 4; ++i) acc[i][j] = wmma_bf16(af[i], bw, acc[i][j]);
    }
    __syncthreads();
  }

  // Scatter to Q/K/V bf16 [B][H][N][HD]; 'which' is constant per block.
  const int hp = lane >> 4;
  const int nn = lane & 15;
  const int which = d0 / C_;
  bf16* outw = (which == 0) ? qws : (which == 1) ? kws : vws;
#pragma unroll
  for (int i = 0; i < 4; ++i)
#pragma unroll
    for (int j = 0; j < 2; ++j)
#pragma unroll
      for (int r = 0; r < 8; ++r) {
        int m   = m0 + wr * 64 + i * 16 + r + 8 * hp;
        int d   = d0 + wc * 32 + j * 16 + nn;
        int cin = d - which * C_;
        int h   = cin >> 6, hd = cin & 63;
        int b   = m >> 11, nseq = m & 2047;
        outw[(((size_t)(b * H_ + h)) * N_ + nseq) * HD_ + hd] =
            f2bf(acc[i][j][r]);
      }
}

// ---------------------------------------------------------------------------
// Kernel 2: fused flash attention. One block per (b*H+h, 128-query tile).
// Q and K tiles staged via GLOBAL_LOAD_ASYNC_TO_LDS_B128 (ASYNCcnt path);
// V is transposed during staging so it stays on the VGPR path.
// Dynamic LDS: Qs 16KB | Ks 16KB | Vs^T 16KB | Ps 8x4KB = 80KB total.
// ---------------------------------------------------------------------------
__global__ __launch_bounds__(256)
void attn_fused(const bf16* __restrict__ qws, const bf16* __restrict__ kws,
                const bf16* __restrict__ vws, bf16* __restrict__ aout) {
  extern __shared__ bf16 smem[];
  bf16* Qs  = smem;                       // [128][64]
  bf16* Ks  = smem + 8192;                // [128][64]
  bf16* Vs  = smem + 16384;               // [64][128] (transposed)
  const int tid  = threadIdx.x;
  const int wave = tid >> 5;
  const int lane = tid & 31;
  bf16* Psw = smem + 24576 + wave * 2048; // per-wave [16][128]

  const int bh = blockIdx.x;              // b*H + h
  const int q0 = blockIdx.y * 128;
  const size_t base = (size_t)bh * N_ * HD_;

  {  // stage Q tile (128x64 bf16 = 16KB) asynchronously
    const char* src = (const char*)(qws + base + (size_t)q0 * HD_);
#pragma unroll
    for (int i = 0; i < 4; ++i) {
      int c = tid + i * 256;              // 1024 x 16B chunks
      async_copy_b128((char*)Qs + c * 16, src + c * 16);
    }
  }

  v8f osum[4] = {};
  float mrun[8], lrun[8];
#pragma unroll
  for (int r = 0; r < 8; ++r) { mrun[r] = -1e30f; lrun[r] = 0.f; }

  for (int kt = 0; kt < N_ / 128; ++kt) {
    __syncthreads();
    {  // stage K tile [key][hd] asynchronously
      const char* src = (const char*)(kws + base + (size_t)kt * 128 * HD_);
#pragma unroll
      for (int i = 0; i < 4; ++i) {
        int c = tid + i * 256;
        async_copy_b128((char*)Ks + c * 16, src + c * 16);
      }
    }
    {  // stage V tile transposed -> Vs[hd][key] (needs VGPR pass)
      const bf16* src = vws + base + (size_t)kt * 128 * HD_;
#pragma unroll
      for (int i = 0; i < 4; ++i) {
        int c   = tid + i * 256;           // 1024 chunks of 8 bf16
        int key = c >> 3;
        int hd0 = (c & 7) * 8;
        uint4 pk = *(const uint4*)(src + key * HD_ + hd0);
        const bf16* e = (const bf16*)&pk;
#pragma unroll
        for (int j = 0; j < 8; ++j) Vs[(hd0 + j) * 128 + key] = e[j];
      }
    }
    async_wait0();          // Q (first iter) + K async copies complete
    __syncthreads();

    // S = Q(16 rows) @ K^T(128 keys): 2 K-steps x 8 N-tiles = 16 WMMAs
    v8f st[8] = {};
#pragma unroll
    for (int ks = 0; ks < 2; ++ks) {
      v16bf aq = frag_a(Qs + (wave * 16) * 64 + ks * 32, 64);
#pragma unroll
      for (int t = 0; t < 8; ++t) {
        v16bf bk = frag_b(Ks + (t * 16) * 64 + ks * 32, 64);
        st[t] = wmma_bf16(aq, bk, st[t]);
      }
    }

    // Online softmax. Row m = r + 8*(lane>>4); its 16 columns per tile live
    // in one 16-lane group, so xor-shuffles 1,2,4,8 reduce exactly that row.
#pragma unroll
    for (int r = 0; r < 8; ++r) {
      float mx = -1e30f;
#pragma unroll
      for (int t = 0; t < 8; ++t) {
        float s = st[t][r] * SCALE_;
        st[t][r] = s;
        mx = fmaxf(mx, s);
      }
#pragma unroll
      for (int off = 1; off < 16; off <<= 1)
        mx = fmaxf(mx, __shfl_xor(mx, off, 32));
      float mnew  = fmaxf(mrun[r], mx);
      float alpha = __expf(mrun[r] - mnew);
      mrun[r] = mnew;
      float psum = 0.f;
      bf16* prow = Psw + (r + 8 * (lane >> 4)) * 128 + (lane & 15);
#pragma unroll
      for (int t = 0; t < 8; ++t) {
        float p = __expf(st[t][r] - mnew);
        psum += p;
        prow[t * 16] = f2bf(p);
      }
#pragma unroll
      for (int off = 1; off < 16; off <<= 1) psum += __shfl_xor(psum, off, 32);
      lrun[r] = lrun[r] * alpha + psum;
#pragma unroll
      for (int j = 0; j < 4; ++j) osum[j][r] *= alpha;
    }

    // O += P @ V : 4 K-steps x 4 N-tiles = 16 WMMAs (Ps is wave-private)
#pragma unroll
    for (int ks = 0; ks < 4; ++ks) {
      v16bf ap = frag_a(Psw + ks * 32, 128);
#pragma unroll
      for (int j = 0; j < 4; ++j) {
        v16bf bv = frag_b(Vs + (j * 16) * 128 + ks * 32, 128);
        osum[j] = wmma_bf16(ap, bv, osum[j]);
      }
    }
  }

  // Write attention output bf16 in [B][N][C] (C index = h*64 + hd).
  const int b = bh >> 4, h = bh & 15;
  const int hp = lane >> 4, nn = lane & 15;
#pragma unroll
  for (int j = 0; j < 4; ++j)
#pragma unroll
    for (int r = 0; r < 8; ++r) {
      int q  = q0 + wave * 16 + r + 8 * hp;
      int hd = j * 16 + nn;
      float o = osum[j][r] / lrun[r];
      aout[((size_t)(b * N_ + q)) * C_ + h * HD_ + hd] = f2bf(o);
    }
}

// ---------------------------------------------------------------------------
// Kernel 3: output projection + bias. aout[8192,1024](bf16) @ w_proj^T + b.
// A tile is a straight bf16 copy -> async global->LDS path.
// ---------------------------------------------------------------------------
__global__ __launch_bounds__(256)
void proj_gemm(const bf16* __restrict__ a, const float* __restrict__ w,
               const float* __restrict__ bias, float* __restrict__ out) {
  __shared__ bf16 As[128 * 32];
  __shared__ bf16 Bs[128 * 32];
  const int tid  = threadIdx.x;
  const int wave = tid >> 5;
  const int lane = tid & 31;
  const int wr = wave >> 2, wc = wave & 3;
  const int m0 = blockIdx.y * 128;
  const int d0 = blockIdx.x * 128;

  v8f acc[4][2] = {};

  for (int k0 = 0; k0 < C_; k0 += 32) {
#pragma unroll
    for (int i = 0; i < 2; ++i) {        // A: async bf16 copy, 512 x 16B
      int c   = tid + i * 256;
      int row = c >> 2;                  // 4 chunks per 64B row
      int col = (c & 3) * 8;
      // LDS byte offset of chunk c in row-major [128][32] bf16 is c*16.
      async_copy_b128((char*)As + c * 16,
                      a + (size_t)(m0 + row) * C_ + k0 + col);
    }
#pragma unroll
    for (int i = 0; i < 4; ++i) {        // B: f32 -> bf16 via VGPRs
      int idx = tid + i * 256;
      int row = idx >> 3;
      int q4  = (idx & 7) * 4;
      float4 vb = *(const float4*)(w + (size_t)(d0 + row) * C_ + k0 + q4);
      bf16* db = Bs + row * 32 + q4;
      db[0] = f2bf(vb.x); db[1] = f2bf(vb.y);
      db[2] = f2bf(vb.z); db[3] = f2bf(vb.w);
    }
    async_wait0();
    __syncthreads();
    v16bf af[4];
#pragma unroll
    for (int i = 0; i < 4; ++i)
      af[i] = frag_a(As + (wr * 64 + i * 16) * 32, 32);
#pragma unroll
    for (int j = 0; j < 2; ++j) {
      v16bf bw = frag_b(Bs + (wc * 32 + j * 16) * 32, 32);
#pragma unroll
      for (int i = 0; i < 4; ++i) acc[i][j] = wmma_bf16(af[i], bw, acc[i][j]);
    }
    __syncthreads();
  }

  const int hp = lane >> 4, nn = lane & 15;
#pragma unroll
  for (int i = 0; i < 4; ++i)
#pragma unroll
    for (int j = 0; j < 2; ++j)
#pragma unroll
      for (int r = 0; r < 8; ++r) {
        int m = m0 + wr * 64 + i * 16 + r + 8 * hp;
        int d = d0 + wc * 32 + j * 16 + nn;
        out[(size_t)m * C_ + d] = acc[i][j][r] + bias[d];
      }
}

extern "C" void kernel_launch(void* const* d_in, const int* in_sizes, int n_in,
                              void* d_out, int out_size, void* d_ws,
                              size_t ws_size, hipStream_t stream) {
  const float* x      = (const float*)d_in[0];
  const float* w_qkv  = (const float*)d_in[1];
  const float* w_proj = (const float*)d_in[2];
  const float* b_proj = (const float*)d_in[3];
  float* out = (float*)d_out;

  const size_t elems = (size_t)B_ * H_ * N_ * HD_;  // 8,388,608
  bf16* qws  = (bf16*)d_ws;
  bf16* kws  = qws + elems;
  bf16* vws  = kws + elems;
  bf16* aout = vws + elems;   // [B][N][C] bf16; total ws use = 64 MB

  dim3 blk(256);
  qkv_gemm<<<dim3(3 * C_ / 128, B_ * N_ / 128), blk, 0, stream>>>(
      x, w_qkv, qws, kws, vws);

  size_t shb = (size_t)(3 * 8192 + 8 * 2048) * sizeof(bf16);  // 81920 B
  attn_fused<<<dim3(B_ * H_, N_ / 128), blk, shb, stream>>>(
      qws, kws, vws, aout);

  proj_gemm<<<dim3(C_ / 128, B_ * N_ / 128), blk, 0, stream>>>(
      aout, w_proj, b_proj, out);
}
